// FP8LinearQwen2DecoderLayer_64261300683335
// MI455X (gfx1250) — 16-bit and 8-bit matrix operand order repaired and checked by reading, compiled, not run on hardware
//
#include <hip/hip_runtime.h>
#include <hip/hip_bf16.h>
#include <stdint.h>

typedef __attribute__((ext_vector_type(16))) int      v16i;
typedef __attribute__((ext_vector_type(8)))  float    v8f;
typedef __attribute__((ext_vector_type(16))) _Float16 v16h;

#define FP8_MAXV 448.0f
#define NHEADS   32
#define HDIM     128

__device__ __forceinline__ void async_copy_b128(const unsigned char* g, void* l) {
  unsigned int lo = (unsigned int)(unsigned long long)(uintptr_t)l;
  unsigned long long ga = (unsigned long long)(uintptr_t)g;
  asm volatile("global_load_async_to_lds_b128 %0, %1, off" : : "v"(lo), "v"(ga) : "memory");
}
__device__ __forceinline__ void wait_async0() {
  asm volatile("s_wait_asynccnt 0x0" : : : "memory");
}

__device__ __forceinline__ unsigned char f32_to_e4m3(float f) {
  unsigned int u = __float_as_uint(f);
  unsigned char sg = (unsigned char)((u >> 31) << 7);
  float a = fabsf(f);
  if (!(a > 0.f)) return sg;
  if (a > FP8_MAXV) a = FP8_MAXV;
  int e;
  float m = frexpf(a, &e);
  int ex = e - 1;
  if (ex < -6) {
    int q = (int)rintf(a * 512.0f);
    if (q >= 8) return sg | 0x08;
    return sg | (unsigned char)q;
  }
  int mant = (int)rintf((m * 2.0f - 1.0f) * 8.0f);
  if (mant == 8) { mant = 0; ++ex; }
  if (ex > 8) { ex = 8; mant = 7; }
  return sg | (unsigned char)((ex + 7) << 3) | (unsigned char)mant;
}

__global__ void k_init(float* p, int n) {
  int i = threadIdx.x;
  if (i < n) p[i] = 0.f;
}

__global__ void k_absmax(const float* __restrict__ x, size_t n, float* amax) {
  __shared__ float red[256];
  float lm = 0.f;
  for (size_t i = (size_t)blockIdx.x * 256u + threadIdx.x; i < n; i += (size_t)gridDim.x * 256u)
    lm = fmaxf(lm, fabsf(x[i]));
  red[threadIdx.x] = lm; __syncthreads();
  for (int s = 128; s > 0; s >>= 1) {
    if ((int)threadIdx.x < s) red[threadIdx.x] = fmaxf(red[threadIdx.x], red[threadIdx.x + s]);
    __syncthreads();
  }
  if (threadIdx.x == 0) atomicMax((int*)amax, __float_as_int(red[0]));
}

__global__ void k_quant(const float* __restrict__ x, size_t n, const float* __restrict__ amax,
                        float* __restrict__ sout, unsigned char* __restrict__ q) {
  float s = amax[0] / FP8_MAXV + 1e-12f;
  if (blockIdx.x == 0 && threadIdx.x == 0) sout[0] = s;
  float inv = 1.0f / s;
  for (size_t i = (size_t)blockIdx.x * 256u + threadIdx.x; i < n; i += (size_t)gridDim.x * 256u)
    q[i] = f32_to_e4m3(x[i] * inv);
}

__global__ void k_rmsnorm_absmax(const float* __restrict__ x, const float* __restrict__ w,
                                 float* __restrict__ y, int Hn, float* amax) {
  __shared__ float red[256];
  const size_t row = blockIdx.x;
  const float* xr = x + row * (size_t)Hn;
  float ss = 0.f;
  for (int i = threadIdx.x; i < Hn; i += 256) { float v = xr[i]; ss += v * v; }
  red[threadIdx.x] = ss; __syncthreads();
  for (int s = 128; s > 0; s >>= 1) {
    if ((int)threadIdx.x < s) red[threadIdx.x] += red[threadIdx.x + s];
    __syncthreads();
  }
  float inv = rsqrtf(red[0] / (float)Hn + 1e-6f);
  __syncthreads();
  float lm = 0.f;
  for (int i = threadIdx.x; i < Hn; i += 256) {
    float v = xr[i] * inv * w[i];
    y[row * (size_t)Hn + i] = v;
    lm = fmaxf(lm, fabsf(v));
  }
  red[threadIdx.x] = lm; __syncthreads();
  for (int s = 128; s > 0; s >>= 1) {
    if ((int)threadIdx.x < s) red[threadIdx.x] = fmaxf(red[threadIdx.x], red[threadIdx.x + s]);
    __syncthreads();
  }
  if (threadIdx.x == 0) atomicMax((int*)amax, __float_as_int(red[0]));
}

__global__ __launch_bounds__(256) void k_gemm_fp8(
    const unsigned char* __restrict__ A, const float* __restrict__ sA,
    const unsigned char* __restrict__ W, const float* __restrict__ sW,
    const float* __restrict__ bias, const float* __restrict__ resid,
    float* __restrict__ out, int M, int N, int K) {
  __shared__ __align__(16) unsigned char At[2][128][128];
  __shared__ __align__(16) unsigned char Bt[2][128][128];
  const int tid = threadIdx.x;
  const int lane = tid & 31, wid = tid >> 5;
  const int lr = lane & 15, hi = (lane >> 4) & 1;
  const int wm = wid & 3, wn = wid >> 2;
  const int m0 = blockIdx.y * 128, n0 = blockIdx.x * 128;
  const int sr = tid >> 1;

  const unsigned char* Ab = A + (size_t)m0 * K;
  const unsigned char* Wb = W + (size_t)n0 * K;

  v8f acc[2][4];
#pragma unroll
  for (int i = 0; i < 2; ++i)
#pragma unroll
    for (int j = 0; j < 4; ++j)
#pragma unroll
      for (int e = 0; e < 8; ++e) acc[i][j][e] = 0.f;

#pragma unroll
  for (int i = 0; i < 4; ++i) {
    int id = i * 256 + tid;
    int r = id >> 3, c = id & 7, cs = c ^ (r & 7);
    async_copy_b128(Ab + (size_t)r * K + c * 16, &At[0][r][cs * 16]);
    async_copy_b128(Wb + (size_t)r * K + c * 16, &Bt[0][r][cs * 16]);
  }

  for (int k0 = 0; k0 < K; k0 += 128) {
    const int cur = (k0 >> 7) & 1;
    wait_async0();
    __syncthreads();

    const int k1 = k0 + 128;
    if (k1 < K) {
#pragma unroll
      for (int i = 0; i < 4; ++i) {
        int id = i * 256 + tid;
        int r = id >> 3, c = id & 7, cs = c ^ (r & 7);
        async_copy_b128(Ab + (size_t)r * K + k1 + c * 16, &At[cur ^ 1][r][cs * 16]);
        async_copy_b128(Wb + (size_t)r * K + k1 + c * 16, &Bt[cur ^ 1][r][cs * 16]);
      }
      if (k1 + 128 < K) {
        __builtin_prefetch(Ab + (size_t)sr * K + k1 + 128, 0, 1);
        __builtin_prefetch(Wb + (size_t)sr * K + k1 + 128, 0, 1);
      }
    }

    v16i af[2], bf[4];
#pragma unroll
    for (int sm = 0; sm < 2; ++sm) {
      const int R = wm * 32 + sm * 16 + lr;
      const unsigned int* pr = (const unsigned int*)&At[cur][R][0];
#pragma unroll
      for (int j = 0; j < 8; ++j) {
        int pj = (((2 * (j >> 1) + hi) ^ (R & 7)) << 2) + 2 * (j & 1);
        af[sm][2 * j]     = (int)pr[pj];
        af[sm][2 * j + 1] = (int)pr[pj + 1];
      }
    }
#pragma unroll
    for (int sn = 0; sn < 4; ++sn) {
      const int R = wn * 64 + sn * 16 + lr;
      const int4* pb4 = (const int4*)&Bt[cur][R][0];
#pragma unroll
      for (int j = 0; j < 4; ++j) {
        int4 d = pb4[(2 * j + hi) ^ (R & 7)];
        bf[sn][4 * j + 0] = d.x;
        bf[sn][4 * j + 1] = d.y;
        bf[sn][4 * j + 2] = d.z;
        bf[sn][4 * j + 3] = d.w;
      }
    }
#pragma unroll
    for (int i = 0; i < 2; ++i)
#pragma unroll
      for (int j = 0; j < 4; ++j)
        acc[i][j] = __builtin_amdgcn_wmma_f32_16x16x128_fp8_fp8(
            af[i], bf[j], (short)0, acc[i][j], false, false);
  }

  const float sc = sA[0] * sW[0];
#pragma unroll
  for (int i = 0; i < 2; ++i)
#pragma unroll
    for (int j = 0; j < 4; ++j) {
      int nn = n0 + wn * 64 + j * 16 + lr;
      float b = bias ? bias[nn] : 0.f;
#pragma unroll
      for (int e = 0; e < 8; ++e) {
        int mm = m0 + wm * 32 + i * 16 + e + 8 * hi;
        size_t idx = (size_t)mm * N + nn;
        float v = acc[i][j][e] * sc + b;
        if (resid) v += resid[idx];
        out[idx] = v;
      }
    }
}

__global__ void k_rope(const float* __restrict__ Q, const float* __restrict__ Kk,
                       const float* __restrict__ V, const int* __restrict__ pos,
                       _Float16* __restrict__ QH, _Float16* __restrict__ KH,
                       _Float16* __restrict__ VH, int S, int Hn) {
  size_t idx = (size_t)blockIdx.x * blockDim.x + threadIdx.x;
  size_t total = (size_t)S * (Hn / 2);
  if (idx >= total) return;
  int s = (int)(idx / (size_t)(Hn / 2));
  int r = (int)(idx % (size_t)(Hn / 2));
  int head = r / 64, j = r % 64;
  int c1 = head * HDIM + j, c2 = c1 + 64;
  float p = (float)pos[s];
  float inv = expf(-((float)j / 64.0f) * 9.210340371976184f);
  float ang = p * inv;
  float cs = cosf(ang), sn = sinf(ang);
  size_t b = (size_t)s * Hn;
  float q1 = Q[b + c1], q2 = Q[b + c2];
  QH[b + c1] = (_Float16)(q1 * cs - q2 * sn);
  QH[b + c2] = (_Float16)(q2 * cs + q1 * sn);
  float k1 = Kk[b + c1], k2 = Kk[b + c2];
  KH[b + c1] = (_Float16)(k1 * cs - k2 * sn);
  KH[b + c2] = (_Float16)(k2 * cs + k1 * sn);
  VH[b + c1] = (_Float16)V[b + c1];
  VH[b + c2] = (_Float16)V[b + c2];
}

__global__ void k_transpose16(const _Float16* __restrict__ in, _Float16* __restrict__ out,
                              int rows, int cols) {
  __shared__ _Float16 tile[32][33];
  int c0 = blockIdx.x * 32, r0 = blockIdx.y * 32;
  for (int i = threadIdx.y; i < 32; i += 8)
    tile[i][threadIdx.x] = in[(size_t)(r0 + i) * cols + c0 + threadIdx.x];
  __syncthreads();
  for (int i = threadIdx.y; i < 32; i += 8)
    out[(size_t)(c0 + i) * rows + r0 + threadIdx.x] = tile[threadIdx.x][i];
}

__global__ __launch_bounds__(32) void k_flash_attn(
    const _Float16* __restrict__ Q, const _Float16* __restrict__ Kc,
    const _Float16* __restrict__ VT, float* __restrict__ out, int S, int Hn) {
  __shared__ _Float16 P[16][32];
  const int lane = threadIdx.x, lr = lane & 15, hi = (lane >> 4) & 1;
  const int q0 = blockIdx.x * 16, head = blockIdx.y;
  const float smul = 0.08838834764831845f;

  const _Float16* qrow = Q + (size_t)(q0 + lr) * Hn + head * HDIM;
  v16h qf[4];
#pragma unroll
  for (int j = 0; j < 4; ++j) {
    const _Float16* c = qrow + j * 32;
#pragma unroll
    for (int u = 0; u < 8; ++u) { qf[j][u] = c[hi * 16 + u]; qf[j][8 + u] = c[hi * 16 + 8 + u]; }
  }

  v8f acc[8];
#pragma unroll
  for (int d = 0; d < 8; ++d)
#pragma unroll
    for (int e = 0; e < 8; ++e) acc[d][e] = 0.f;
  float m[8], l[8];
#pragma unroll
  for (int e = 0; e < 8; ++e) { m[e] = -3.0e38f; l[e] = 0.f; }

  for (int kc = 0; kc < q0 + 16; kc += 32) {
    v8f st[2];
#pragma unroll
    for (int t = 0; t < 2; ++t) {
      const _Float16* krow = Kc + (size_t)(kc + t * 16 + lr) * Hn + head * HDIM;
      v8f s;
#pragma unroll
      for (int e = 0; e < 8; ++e) s[e] = 0.f;
#pragma unroll
      for (int j = 0; j < 4; ++j) {
        v16h bfk = *(const v16h*)(krow + j * 32 + hi * 16);
        s = __builtin_amdgcn_wmma_f32_16x16x32_f16(false, qf[j], false, bfk, (short)0, s, false, false);
      }
      st[t] = s;
    }
    float fac[8];
#pragma unroll
    for (int e = 0; e < 8; ++e) {
      int row = q0 + e + 8 * hi;
#pragma unroll
      for (int t = 0; t < 2; ++t) {
        int col = kc + t * 16 + lr;
        float v = st[t][e] * smul;
        if (col > row) v = -3.0e38f;
        st[t][e] = v;
      }
      float mx = fmaxf(st[0][e], st[1][e]);
      mx = fmaxf(mx, __shfl_xor(mx, 1, 16));
      mx = fmaxf(mx, __shfl_xor(mx, 2, 16));
      mx = fmaxf(mx, __shfl_xor(mx, 4, 16));
      mx = fmaxf(mx, __shfl_xor(mx, 8, 16));
      float mn = fmaxf(m[e], mx);
      fac[e] = __expf(m[e] - mn);
      m[e] = mn;
      float p0 = __expf(st[0][e] - mn), p1 = __expf(st[1][e] - mn);
      st[0][e] = p0; st[1][e] = p1;
      float rs = p0 + p1;
      rs += __shfl_xor(rs, 1, 16);
      rs += __shfl_xor(rs, 2, 16);
      rs += __shfl_xor(rs, 4, 16);
      rs += __shfl_xor(rs, 8, 16);
      l[e] = l[e] * fac[e] + rs;
    }
#pragma unroll
    for (int d = 0; d < 8; ++d)
#pragma unroll
      for (int e = 0; e < 8; ++e) acc[d][e] *= fac[e];

#pragma unroll
    for (int e = 0; e < 8; ++e) {
      P[e + 8 * hi][lr]      = (_Float16)st[0][e];
      P[e + 8 * hi][16 + lr] = (_Float16)st[1][e];
    }
    __syncthreads();
    v16h pf;
#pragma unroll
    for (int u = 0; u < 8; ++u) { pf[u] = P[lr][hi * 16 + u]; pf[8 + u] = P[lr][hi * 16 + 8 + u]; }
    __syncthreads();

#pragma unroll
    for (int d = 0; d < 8; ++d) {
      const _Float16* vrow = VT + (size_t)(head * HDIM + d * 16 + lr) * S + kc + hi * 16;
      v16h vb = *(const v16h*)vrow;
      acc[d] = __builtin_amdgcn_wmma_f32_16x16x32_f16(false, pf, false, vb, (short)0, acc[d], false, false);
    }
  }
#pragma unroll
  for (int d = 0; d < 8; ++d)
#pragma unroll
    for (int e = 0; e < 8; ++e) {
      size_t o = (size_t)(q0 + e + 8 * hi) * Hn + head * HDIM + d * 16 + lr;
      out[o] = acc[d][e] / l[e];
    }
}

__global__ void k_silu_mul_absmax(const float* __restrict__ g, const float* __restrict__ u,
                                  float* __restrict__ o, size_t n, float* amax) {
  __shared__ float red[256];
  float lm = 0.f;
  for (size_t i = (size_t)blockIdx.x * 256u + threadIdx.x; i < n; i += (size_t)gridDim.x * 256u) {
    float x = g[i];
    float t = (x / (1.f + __expf(-x))) * u[i];
    o[i] = t;
    lm = fmaxf(lm, fabsf(t));
  }
  red[threadIdx.x] = lm; __syncthreads();
  for (int s = 128; s > 0; s >>= 1) {
    if ((int)threadIdx.x < s) red[threadIdx.x] = fmaxf(red[threadIdx.x], red[threadIdx.x + s]);
    __syncthreads();
  }
  if (threadIdx.x == 0) atomicMax((int*)amax, __float_as_int(red[0]));
}

extern "C" void kernel_launch(void* const* d_in, const int* in_sizes, int n_in,
                              void* d_out, int out_size, void* d_ws, size_t ws_size,
                              hipStream_t stream) {
  const float* x    = (const float*)d_in[0];
  const int*   pos  = (const int*)d_in[1];
  const float* ln1w = (const float*)d_in[2];
  const float* wq   = (const float*)d_in[3];
  const float* bq   = (const float*)d_in[4];
  const float* wk   = (const float*)d_in[5];
  const float* bk   = (const float*)d_in[6];
  const float* wv   = (const float*)d_in[7];
  const float* bv   = (const float*)d_in[8];
  const float* wo   = (const float*)d_in[9];
  const float* wg   = (const float*)d_in[10];
  const float* wu   = (const float*)d_in[11];
  const float* wd   = (const float*)d_in[12];
  const float* ln2w = (const float*)d_in[13];
  float* out = (float*)d_out;

  const int H = in_sizes[2];
  const int S = in_sizes[0] / H;
  const int I = in_sizes[10] / H;

  char* ws = (char*)d_ws;
  size_t off = 0;
  auto alloc = [&](size_t bytes) -> char* {
    char* p = ws + off;
    off += (bytes + 255) & ~(size_t)255;
    return p;
  };

  float* amax = (float*)alloc(64 * sizeof(float));
  float* scl  = amax + 16;
  float* Y32  = (float*)alloc((size_t)S * H * 4);
  unsigned char* QX1 = (unsigned char*)alloc((size_t)S * H);
  unsigned char* QX2 = (unsigned char*)alloc((size_t)S * H);
  float* Q32 = (float*)alloc((size_t)S * H * 4);
  float* K32 = (float*)alloc((size_t)S * H * 4);
  float* V32 = (float*)alloc((size_t)S * H * 4);
  _Float16* QH = (_Float16*)alloc((size_t)S * H * 2);
  _Float16* KH = (_Float16*)alloc((size_t)S * H * 2);
  _Float16* VH = (_Float16*)alloc((size_t)S * H * 2);
  _Float16* VT = (_Float16*)alloc((size_t)H * S * 2);
  unsigned char* W8q = (unsigned char*)alloc((size_t)H * H);
  unsigned char* W8k = (unsigned char*)alloc((size_t)H * H);
  unsigned char* W8v = (unsigned char*)alloc((size_t)H * H);
  unsigned char* W8o = (unsigned char*)alloc((size_t)H * H);
  unsigned char* WG8 = (unsigned char*)alloc((size_t)I * H);
  unsigned char* WU8 = (unsigned char*)alloc((size_t)I * H);
  unsigned char* WD8 = (unsigned char*)alloc((size_t)H * I);
  float* G32 = (float*)alloc((size_t)S * I * 4);
  float* U32 = (float*)alloc((size_t)S * I * 4);
  unsigned char* P8 = (unsigned char*)alloc((size_t)S * I);

  k_init<<<1, 64, 0, stream>>>(amax, 16);

  k_rmsnorm_absmax<<<S, 256, 0, stream>>>(x, ln1w, Y32, H, amax + 0);
  k_absmax<<<1024, 256, 0, stream>>>(wq, (size_t)H * H, amax + 1);
  k_absmax<<<1024, 256, 0, stream>>>(wk, (size_t)H * H, amax + 2);
  k_absmax<<<1024, 256, 0, stream>>>(wv, (size_t)H * H, amax + 3);
  k_absmax<<<1024, 256, 0, stream>>>(wo, (size_t)H * H, amax + 4);
  k_absmax<<<4096, 256, 0, stream>>>(wg, (size_t)I * H, amax + 7);
  k_absmax<<<4096, 256, 0, stream>>>(wu, (size_t)I * H, amax + 8);
  k_absmax<<<4096, 256, 0, stream>>>(wd, (size_t)H * I, amax + 9);

  k_quant<<<1024, 256, 0, stream>>>(Y32, (size_t)S * H, amax + 0, scl + 0, QX1);
  k_quant<<<1024, 256, 0, stream>>>(wq, (size_t)H * H, amax + 1, scl + 1, W8q);
  k_quant<<<1024, 256, 0, stream>>>(wk, (size_t)H * H, amax + 2, scl + 2, W8k);
  k_quant<<<1024, 256, 0, stream>>>(wv, (size_t)H * H, amax + 3, scl + 3, W8v);
  k_quant<<<1024, 256, 0, stream>>>(wo, (size_t)H * H, amax + 4, scl + 4, W8o);
  k_quant<<<4096, 256, 0, stream>>>(wg, (size_t)I * H, amax + 7, scl + 7, WG8);
  k_quant<<<4096, 256, 0, stream>>>(wu, (size_t)I * H, amax + 8, scl + 8, WU8);
  k_quant<<<4096, 256, 0, stream>>>(wd, (size_t)H * I, amax + 9, scl + 9, WD8);

  dim3 gq(H / 128, S / 128);
  k_gemm_fp8<<<gq, 256, 0, stream>>>(QX1, scl + 0, W8q, scl + 1, bq, nullptr, Q32, S, H, H);
  k_gemm_fp8<<<gq, 256, 0, stream>>>(QX1, scl + 0, W8k, scl + 2, bk, nullptr, K32, S, H, H);
  k_gemm_fp8<<<gq, 256, 0, stream>>>(QX1, scl + 0, W8v, scl + 3, bv, nullptr, V32, S, H, H);

  size_t nrope = (size_t)S * (H / 2);
  k_rope<<<(unsigned)((nrope + 255) / 256), 256, 0, stream>>>(Q32, K32, V32, pos, QH, KH, VH, S, H);
  dim3 gt(H / 32, S / 32);
  k_transpose16<<<gt, dim3(32, 8), 0, stream>>>(VH, VT, S, H);

  dim3 gf(S / 16, NHEADS);
  k_flash_attn<<<gf, 32, 0, stream>>>(QH, KH, VT, Y32, S, H);

  k_absmax<<<1024, 256, 0, stream>>>(Y32, (size_t)S * H, amax + 5);
  k_quant<<<1024, 256, 0, stream>>>(Y32, (size_t)S * H, amax + 5, scl + 5, QX2);
  k_gemm_fp8<<<gq, 256, 0, stream>>>(QX2, scl + 5, W8o, scl + 4, nullptr, x, K32, S, H, H);

  k_rmsnorm_absmax<<<S, 256, 0, stream>>>(K32, ln2w, Y32, H, amax + 6);
  k_quant<<<1024, 256, 0, stream>>>(Y32, (size_t)S * H, amax + 6, scl + 6, QX1);

  dim3 gg(I / 128, S / 128);
  k_gemm_fp8<<<gg, 256, 0, stream>>>(QX1, scl + 6, WG8, scl + 7, nullptr, nullptr, G32, S, I, H);
  k_gemm_fp8<<<gg, 256, 0, stream>>>(QX1, scl + 6, WU8, scl + 8, nullptr, nullptr, U32, S, I, H);

  k_silu_mul_absmax<<<4096, 256, 0, stream>>>(G32, U32, G32, (size_t)S * I, amax + 10);
  k_quant<<<4096, 256, 0, stream>>>(G32, (size_t)S * I, amax + 10, scl + 10, P8);

  k_gemm_fp8<<<gq, 256, 0, stream>>>(P8, scl + 10, WD8, scl + 9, nullptr, K32, out, S, H, I);
}
